// _SpatialAttention_49134425866850
// MI455X (gfx1250) — matrix operand order repaired and checked by reading, not compiled, not run on hardware
//
#include <hip/hip_runtime.h>
#include <hip/hip_bf16.h>
#include <math.h>

#define N_TOK 8192
#define DMODEL 512
#define HDIM 512

typedef __bf16 bf16_t;
typedef __attribute__((ext_vector_type(16))) __bf16 v16bf;
typedef __attribute__((ext_vector_type(8)))  __bf16 v8bf;
typedef __attribute__((ext_vector_type(8)))  float  v8f;

typedef __attribute__((ext_vector_type(4))) unsigned int u32x4;
typedef __attribute__((ext_vector_type(8))) int          i32x8;
typedef __attribute__((ext_vector_type(4))) int          i32x4;

#if defined(__has_builtin)
#  if __has_builtin(__builtin_amdgcn_tensor_load_to_lds) && \
      __has_builtin(__builtin_amdgcn_s_wait_tensorcnt)
#    define USE_TDM 1
#  endif
#endif
#ifndef USE_TDM
#  define USE_TDM 0
#endif

static __device__ inline bf16_t f2bf(float f) {
    union { float f; unsigned u; } x; x.f = f;
    unsigned r = (x.u + 0x7FFFu + ((x.u >> 16) & 1u)) >> 16;
    unsigned short h = (unsigned short)r;
    bf16_t b;
    __builtin_memcpy(&b, &h, 2);
    return b;
}

static __device__ inline v16bf cat16(v8bf lo, v8bf hi) {
    return __builtin_shufflevector(lo, hi, 0,1,2,3,4,5,6,7,8,9,10,11,12,13,14,15);
}

static __device__ inline v8f wmma_bf16(v16bf a, v16bf b, v8f c) {
    return __builtin_amdgcn_wmma_f32_16x16x32_bf16(
        false, a, false, b, (short)0, c, false, false);
}

#if USE_TDM
// 1-D contiguous TDM load: nelem 2-byte elements from global -> LDS offset.
// D# layout per CDNA5 ISA 8.3/8.4 (group0: count/lds/global/type, group1:
// data_size=2B, tensor_dim0 = tile_dim0 = nelem, tensor_dim1 = tile_dim1 = 1).
// This toolchain exposes the 6-argument builtin (4+8+4+4+8 dwords + cpol).
static __device__ __attribute__((always_inline)) void tdm_load_1d(
        unsigned lds_addr, const void* gptr, unsigned nelem) {
    unsigned long long ga = (unsigned long long)gptr;
    u32x4 g0;
    g0[0] = 1u;                                           // count=1, user mode
    g0[1] = lds_addr;                                     // LDS byte address
    g0[2] = (unsigned)(ga & 0xFFFFFFFFu);                 // global_addr[31:0]
    g0[3] = (unsigned)((ga >> 32) & 0x01FFFFFFu) | (2u << 30); // [56:32] | type=2
    i32x8 g1;
    g1[0] = (int)(1u << 16);                              // data_size=1 (2 bytes)
    g1[1] = (int)((nelem & 0xFFFFu) << 16);               // tensor_dim0[15:0]
    g1[2] = (int)((nelem >> 16) | (1u << 16));            // dim0[31:16] | tensor_dim1=1
    g1[3] = (int)((nelem & 0xFFFFu) << 16);               // tile_dim0 = nelem
    g1[4] = (int)1;                                       // tile_dim1=1, tile_dim2=0
    g1[5] = (int)nelem;                                   // tensor_dim0_stride
    g1[6] = 0;
    g1[7] = 0;
    i32x4 z4 = {0, 0, 0, 0};
    i32x8 z8 = {0, 0, 0, 0, 0, 0, 0, 0};
    __builtin_amdgcn_tensor_load_to_lds(g0, g1, z4, z4, z8, 0);
}
#endif

// ---------------------------------------------------------------------------
// Stage 0a: fp32 -> bf16 elementwise convert (x)
// ---------------------------------------------------------------------------
__global__ void k_cvt_bf16(const float* __restrict__ src, bf16_t* __restrict__ dst, int n) {
    int i = blockIdx.x * blockDim.x + threadIdx.x;
    if (i < n) dst[i] = f2bf(src[i]);
}

// ---------------------------------------------------------------------------
// Stage 0b: repack a 512x512 fp32 weight (row-major [K][N]) into the WMMA
// B-fragment lane layout: P[kt][nt][l][j] = W[32*kt + 16*(l>>4) + j][16*nt + (l&15)]
// ---------------------------------------------------------------------------
__global__ void k_pack_w(const float* __restrict__ W, bf16_t* __restrict__ P) {
    int idx = blockIdx.x * blockDim.x + threadIdx.x;      // 0 .. 512*512-1
    int j  = idx & 15;
    int l  = (idx >> 4) & 31;
    int nt = (idx >> 9) & 31;
    int kt = idx >> 14;
    int k  = 32 * kt + 16 * (l >> 4) + j;
    int n  = 16 * nt + (l & 15);
    P[idx] = f2bf(W[k * HDIM + n]);
}

// ---------------------------------------------------------------------------
// Stage 1: projections.  One wave computes one 16x16 output tile of x @ W.
// mode 0: row-major bf16 [N][H] (Q);  mode 1: packed B-frag for K^T;
// mode 2: +bias, packed B-frag for V.
// ---------------------------------------------------------------------------
__global__ void k_proj(const bf16_t* __restrict__ xb, const bf16_t* __restrict__ WP,
                       const float* __restrict__ bias, bf16_t* __restrict__ out, int mode) {
    int wave = threadIdx.x >> 5;
    int lane = threadIdx.x & 31;
    int tile = blockIdx.x * 8 + wave;
    int mt = tile >> 5;
    int nt = tile & 31;
    int row  = lane & 15;
    int koff = (lane < 16) ? 0 : 16;
    const bf16_t* arow = xb + (size_t)(mt * 16 + row) * DMODEL;

    v8f c = {0.f, 0.f, 0.f, 0.f, 0.f, 0.f, 0.f, 0.f};
    for (int ks = 0; ks < DMODEL / 32; ++ks) {
        int k0 = ks * 32;
        v16bf a = cat16(*(const v8bf*)(arow + k0 + koff),
                        *(const v8bf*)(arow + k0 + 8 + koff));
        const bf16_t* bp = WP + ((size_t)(ks * 32 + nt) * 32 + lane) * 16;
        v16bf b = cat16(*(const v8bf*)bp, *(const v8bf*)(bp + 8));
        c = wmma_bf16(a, b, c);
    }

    int hcol  = nt * 16 + (lane & 15);
    int rbase = mt * 16 + ((lane < 16) ? 0 : 8);
    if (mode == 0) {
        #pragma unroll
        for (int v = 0; v < 8; ++v)
            out[(size_t)(rbase + v) * HDIM + hcol] = f2bf(c[v]);
    } else if (mode == 1) {
        int kt = hcol >> 5;
        int lhi = ((hcol >> 4) & 1) << 4;
        int j = hcol & 15;
        #pragma unroll
        for (int v = 0; v < 8; ++v) {
            int n = rbase + v;
            int l = (n & 15) | lhi;
            out[(((size_t)kt * (N_TOK / 16) + (n >> 4)) * 32 + l) * 16 + j] = f2bf(c[v]);
        }
    } else {
        float bvv = bias ? bias[hcol] : 0.f;
        #pragma unroll
        for (int v = 0; v < 8; ++v) {
            int n = rbase + v;
            int kt = n >> 5;
            int l = (hcol & 15) | (((n >> 4) & 1) << 4);
            int j = n & 15;
            out[(((size_t)kt * (DMODEL / 16) + (hcol >> 4)) * 32 + l) * 16 + j] =
                f2bf(c[v] + bvv);
        }
    }
}

// ---------------------------------------------------------------------------
// Stage 2: dual-branch scores + exp + mask + row sums.
// Block = 16-row strip; Q strips staged in LDS by TDM (tensor_load_to_lds).
// Each wave owns a 16x64 strip (4 m-tiles): A-frags loaded once per K-step
// and reused across 4 B tiles -> 8 WMMAs per 4 ds_load_b128.
// ---------------------------------------------------------------------------
__global__ void __launch_bounds__(256)
k_scores(const bf16_t* __restrict__ Qr, const bf16_t* __restrict__ Qf,
         const bf16_t* __restrict__ KrP, const bf16_t* __restrict__ KfP,
         const unsigned char* __restrict__ mr, const unsigned char* __restrict__ mf,
         float* __restrict__ attn, float* __restrict__ rowsum) {
    // single LDS object -> base offset 0: [0,16K) sQr, [16K,32K) sQf, rs at 32K
    __shared__ __align__(16) char smem[16 * HDIM * 2 * 2 + 64];
    bf16_t* sQr = (bf16_t*)smem;
    bf16_t* sQf = (bf16_t*)(smem + 16 * HDIM * 2);
    float*  rs  = (float*)(smem + 16 * HDIM * 2 * 2);

    int tid = threadIdx.x;
    int r0  = blockIdx.x * 16;

#if USE_TDM
    if (tid < 32) {   // wave 0 drives the DMA; others wait at the barrier
        tdm_load_1d(0u, Qr + (size_t)r0 * HDIM, 16 * HDIM);
        tdm_load_1d((unsigned)(16 * HDIM * 2), Qf + (size_t)r0 * HDIM, 16 * HDIM);
        __builtin_amdgcn_s_wait_tensorcnt(0);
    }
#else
    {
        const uint4* gr = (const uint4*)(Qr + (size_t)r0 * HDIM);
        const uint4* gf = (const uint4*)(Qf + (size_t)r0 * HDIM);
        uint4* sr4 = (uint4*)sQr;
        uint4* sf4 = (uint4*)sQf;
        for (int i = tid; i < 16 * HDIM * 2 / 16; i += 256) { sr4[i] = gr[i]; sf4[i] = gf[i]; }
    }
#endif
    if (tid < 16) rs[tid] = 0.f;
    __syncthreads();

    int wave = tid >> 5, lane = tid & 31;
    int row  = lane & 15;
    int koff = (lane < 16) ? 0 : 16;
    int rsel = (lane < 16) ? 0 : 8;
    const float inv_denom = 1.0f / 1.5f;
    const float scale = 0.044194173824159216f;   // 1/sqrt(512)

    float part[8];
    #pragma unroll
    for (int v = 0; v < 8; ++v) part[v] = 0.f;

    for (int outer = 0; outer < (N_TOK / 16) / 32; ++outer) {   // 16 outer passes
        int mbase = outer * 32 + wave * 4;                      // 4 tiles per wave
        v8f cr[4], cf[4];
        #pragma unroll
        for (int q = 0; q < 4; ++q) {
            v8f z = {0.f, 0.f, 0.f, 0.f, 0.f, 0.f, 0.f, 0.f};
            cr[q] = z; cf[q] = z;
        }
        for (int ks = 0; ks < HDIM / 32; ++ks) {
            int k0 = ks * 32;
            v16bf ar = cat16(*(const v8bf*)&sQr[row * HDIM + k0 + koff],
                             *(const v8bf*)&sQr[row * HDIM + k0 + 8 + koff]);
            v16bf af = cat16(*(const v8bf*)&sQf[row * HDIM + k0 + koff],
                             *(const v8bf*)&sQf[row * HDIM + k0 + 8 + koff]);
            #pragma unroll
            for (int q = 0; q < 4; ++q) {
                const bf16_t* bpr =
                    KrP + ((size_t)(ks * (N_TOK / 16) + mbase + q) * 32 + lane) * 16;
                const bf16_t* bpf =
                    KfP + ((size_t)(ks * (N_TOK / 16) + mbase + q) * 32 + lane) * 16;
                v16bf br = cat16(*(const v8bf*)bpr, *(const v8bf*)(bpr + 8));
                v16bf bb = cat16(*(const v8bf*)bpf, *(const v8bf*)(bpf + 8));
                cr[q] = wmma_bf16(ar, br, cr[q]);
                cf[q] = wmma_bf16(af, bb, cf[q]);
            }
        }
        #pragma unroll
        for (int q = 0; q < 4; ++q) {
            int col = (mbase + q) * 16 + (lane & 15);
            #pragma unroll
            for (int v = 0; v < 8; ++v) {
                size_t off = (size_t)(r0 + rsel + v) * N_TOK + col;
                unsigned char mrv = __builtin_nontemporal_load(mr + off);
                unsigned char mfv = __builtin_nontemporal_load(mf + off);
                float er = mrv ? 0.f : __expf(cr[q][v] * scale);
                float ef = mfv ? 0.f : __expf(cf[q][v] * scale);
                float comb = (er + 0.5f * ef) * inv_denom;
                part[v] += comb;
                __builtin_nontemporal_store(comb, attn + off);  // stage 3 normalizes
            }
        }
    }
    #pragma unroll
    for (int v = 0; v < 8; ++v) atomicAdd(&rs[rsel + v], part[v]);
    __syncthreads();
    if (tid < 16) rowsum[r0 + tid] = rs[tid];
}

// ---------------------------------------------------------------------------
// Stage 3: normalize attn in place + cntx = attn @ V (bf16 WMMA, fp32 acc).
// Double-buffered LDS A-strip: one barrier per K-step.
// ---------------------------------------------------------------------------
__global__ void __launch_bounds__(256)
k_attnv(float* __restrict__ attn, const float* __restrict__ rowsum,
        const bf16_t* __restrict__ VP, float* __restrict__ cntx) {
    __shared__ bf16_t sA[2][16 * 32];
    __shared__ float  sInv[16];

    int tid = threadIdx.x;
    int r0  = blockIdx.x * 16;
    if (tid < 16) sInv[tid] = 1.0f / rowsum[r0 + tid];
    __syncthreads();

    int wave = tid >> 5, lane = tid & 31;
    int row  = lane & 15;
    int koff = (lane < 16) ? 0 : 16;

    v8f c[4];
    #pragma unroll
    for (int q = 0; q < 4; ++q) {
        v8f z = {0.f, 0.f, 0.f, 0.f, 0.f, 0.f, 0.f, 0.f};
        c[q] = z;
    }

    for (int ms = 0; ms < N_TOK / 32; ++ms) {
        int pb = ms & 1;
        #pragma unroll
        for (int t = 0; t < 2; ++t) {
            int idx = tid + t * 256;                     // 0..511
            int rr = idx >> 5, cc = idx & 31;
            size_t off = (size_t)(r0 + rr) * N_TOK + (size_t)ms * 32 + cc;
            float v = __builtin_nontemporal_load(attn + off) * sInv[rr];
            __builtin_nontemporal_store(v, attn + off);  // final normalized attn
            sA[pb][rr * 32 + cc] = f2bf(v);
        }
        __syncthreads();

        v16bf a = cat16(*(const v8bf*)&sA[pb][row * 32 + koff],
                        *(const v8bf*)&sA[pb][row * 32 + 8 + koff]);
        #pragma unroll
        for (int q = 0; q < 4; ++q) {
            int dt = wave * 4 + q;
            const bf16_t* bp = VP + ((size_t)(ms * (DMODEL / 16) + dt) * 32 + lane) * 16;
            v16bf b = cat16(*(const v8bf*)bp, *(const v8bf*)(bp + 8));
            c[q] = wmma_bf16(a, b, c[q]);
        }
    }

    int rsel = (lane < 16) ? 0 : 8;
    #pragma unroll
    for (int q = 0; q < 4; ++q) {
        int dcol = (wave * 4 + q) * 16 + (lane & 15);
        #pragma unroll
        for (int v = 0; v < 8; ++v)
            cntx[(size_t)(r0 + rsel + v) * DMODEL + dcol] = c[q][v];
    }
}

// ---------------------------------------------------------------------------
extern "C" void kernel_launch(void* const* d_in, const int* in_sizes, int n_in,
                              void* d_out, int out_size, void* d_ws, size_t ws_size,
                              hipStream_t stream) {
    const float* x = (const float*)d_in[0];
    const unsigned char* mr = (const unsigned char*)d_in[1];
    const unsigned char* mf = (const unsigned char*)d_in[2];
    const float* Wqr = (const float*)d_in[3];
    const float* Wkr = (const float*)d_in[4];
    const float* Wqf = (const float*)d_in[5];
    const float* Wkf = (const float*)d_in[6];
    const float* Wv  = (const float*)d_in[7];
    const float* bv  = (const float*)d_in[8];

    char* ws = (char*)d_ws;
    size_t off = 0;
    auto alloc = [&](size_t bytes) {
        void* p = ws + off;
        off += (bytes + 255) & ~(size_t)255;
        return p;
    };
    bf16_t* xb   = (bf16_t*)alloc((size_t)N_TOK * DMODEL * 2);
    bf16_t* Qr   = (bf16_t*)alloc((size_t)N_TOK * HDIM * 2);
    bf16_t* Qf   = (bf16_t*)alloc((size_t)N_TOK * HDIM * 2);
    bf16_t* KrP  = (bf16_t*)alloc((size_t)N_TOK * HDIM * 2);
    bf16_t* KfP  = (bf16_t*)alloc((size_t)N_TOK * HDIM * 2);
    bf16_t* VP   = (bf16_t*)alloc((size_t)N_TOK * DMODEL * 2);
    bf16_t* WqrP = (bf16_t*)alloc((size_t)DMODEL * HDIM * 2);
    bf16_t* WkrP = (bf16_t*)alloc((size_t)DMODEL * HDIM * 2);
    bf16_t* WqfP = (bf16_t*)alloc((size_t)DMODEL * HDIM * 2);
    bf16_t* WkfP = (bf16_t*)alloc((size_t)DMODEL * HDIM * 2);
    bf16_t* WvP  = (bf16_t*)alloc((size_t)DMODEL * HDIM * 2);
    float*  rowsum = (float*)alloc((size_t)N_TOK * 4);

    k_cvt_bf16<<<(N_TOK * DMODEL + 255) / 256, 256, 0, stream>>>(x, xb, N_TOK * DMODEL);
    k_pack_w<<<(DMODEL * HDIM) / 256, 256, 0, stream>>>(Wqr, WqrP);
    k_pack_w<<<(DMODEL * HDIM) / 256, 256, 0, stream>>>(Wkr, WkrP);
    k_pack_w<<<(DMODEL * HDIM) / 256, 256, 0, stream>>>(Wqf, WqfP);
    k_pack_w<<<(DMODEL * HDIM) / 256, 256, 0, stream>>>(Wkf, WkfP);
    k_pack_w<<<(DMODEL * HDIM) / 256, 256, 0, stream>>>(Wv, WvP);

    int proj_blocks = (N_TOK / 16) * (HDIM / 16) / 8;
    k_proj<<<proj_blocks, 256, 0, stream>>>(xb, WqrP, nullptr, Qr, 0);
    k_proj<<<proj_blocks, 256, 0, stream>>>(xb, WkrP, nullptr, KrP, 1);
    k_proj<<<proj_blocks, 256, 0, stream>>>(xb, WqfP, nullptr, Qf, 0);
    k_proj<<<proj_blocks, 256, 0, stream>>>(xb, WkfP, nullptr, KfP, 1);
    k_proj<<<proj_blocks, 256, 0, stream>>>(xb, WvP, bv, VP, 2);

    float* cntx = (float*)d_out;
    float* attn = (float*)d_out + (size_t)N_TOK * DMODEL;

    k_scores<<<N_TOK / 16, 256, 0, stream>>>(Qr, Qf, KrP, KfP, mr, mf, attn, rowsum);
    k_attnv<<<N_TOK / 16, 256, 0, stream>>>(attn, rowsum, VP, cntx);
}
